// DVAE_42992622633003
// MI455X (gfx1250) — compile-verified, matrix operand order checked by reading, not run on hardware
//
#include <hip/hip_runtime.h>
#include <hip/hip_bf16.h>

// ---------------------------------------------------------------------------
// DVAE forward for MI455X (gfx1250, wave32, WMMA).
//   - GRU scans: 16 persistent workgroups (2 GRUs x 8 batch-groups of 16 rows),
//     weights resident in VGPRs as WMMA fragments, h in LDS (f16) + registers,
//     branch-free inner loop (t=0 peeled), V_TANH_F32 gate activations,
//     x pre-packed to f16 (no converts on the serial path).
//   - MLPs: f16 WMMA, weights staged in LDS (>64KB -> dynamic shared).
//   - KL: deterministic two-stage fixed-order reduction (no float atomics).
// ---------------------------------------------------------------------------

#define T_DIM  2048
#define B_DIM  128
#define X_DIMC 64
#define U_DIMC 16
#define Z_DIMC 32
#define H_DIMC 128
#define MLP_HC 128

typedef __attribute__((ext_vector_type(16))) _Float16     v16h;
typedef __attribute__((ext_vector_type(8)))  _Float16     v8h;
typedef __attribute__((ext_vector_type(4)))  _Float16     v4h;
typedef __attribute__((ext_vector_type(8)))  float        v8f;
typedef __attribute__((ext_vector_type(4)))  float        v4f;
typedef __attribute__((ext_vector_type(4)))  unsigned int v4u;

// ---- f16 weight-pack layout in workspace (offsets in halves) ----
#define OFF_DH_WIH 0        // 384 x 64
#define OFF_DH_WHH 24576    // 384 x 128
#define OFF_EH_WIH 73728
#define OFF_EH_WHH 98304
#define OFF_MLP    147456   // contiguous MLP weight block starts here
// offsets inside the MLP block (also the LDS carve offsets, in halves)
#define MO_W1P 0            // 128 x 160   (cols 144..159 zero-padded)
#define MO_W2P 20480        // 64  x 128
#define MO_W1Q 28672        // 128 x 160
#define MO_W2Q 49152        // 64  x 128
#define MO_W1D 57344        // 128 x 160   (h128 | z32 -> exact)
#define MO_W2D 77824        // 128 x 128
#define MLP_W_HALVES 94208
#define PACK_TOTAL   241664

// workspace byte offsets
#define WS_HSP_OFF  (512u * 1024u)
#define HS_BYTES    ((size_t)T_DIM * B_DIM * H_DIMC * 2)          // 64 MB
#define WS_HSQ_OFF  ((size_t)WS_HSP_OFF + HS_BYTES)
#define WS_KLP_OFF  ((size_t)WS_HSP_OFF + 2 * HS_BYTES)           // 2048 f32
#define WS_XH_OFF   ((size_t)WS_KLP_OFF + 8192)                   // x in f16, 32MB
#define X_ELEMS     ((size_t)B_DIM * T_DIM * X_DIMC)              // 16,777,216
#define MLP_BLOCKS  ((B_DIM * T_DIM) / 128)                       // 2048

// mlp-kernel LDS carve (halves), after the 6 staged weight matrices
#define SM_STAGE    94208                       // 8 waves * 16*136
#define SM_KL_HALF  (SM_STAGE + 8 * 16 * 136)   // 111616 halves
#define MLP_SMEM_BYTES (SM_KL_HALF * 2 + 256 * 4)   // 224256 B

// ---------------------------------------------------------------------------
// WMMA + fragment helpers (16x16x32 f16 -> f32 accumulate)
// ---------------------------------------------------------------------------
__device__ __forceinline__ v8f wmma(v16h a, v16h b, v8f c) {
  return __builtin_amdgcn_wmma_f32_16x16x32_f16(false, a, false, b, (short)0, c,
                                                false, false);
}

// 16x32 f16 fragment: lane (l&15) = row/out-col, ks {kb..kb+7, 16+kb..16+kb+7},
// kb = (l&16)?8:0. Both chunks are contiguous 16B -> two b128 loads.
__device__ __forceinline__ v16h frag_ld16h(const _Float16* __restrict__ src,
                                           int ld, int r0, int kc, int lane) {
  const _Float16* p =
      src + (size_t)(r0 + (lane & 15)) * ld + kc + ((lane & 16) ? 8 : 0);
  v8h lo = *(const v8h*)p;
  v8h hi = *(const v8h*)(p + 16);
  return __builtin_shufflevector(lo, hi, 0, 1, 2, 3, 4, 5, 6, 7, 8, 9, 10, 11,
                                 12, 13, 14, 15);
}

__device__ __forceinline__ v16h cvt16(v4f a, v4f b, v4f c, v4f d) {
  v16h f;
#pragma unroll
  for (int i = 0; i < 4; ++i) {
    f[i]      = (_Float16)a[i];
    f[4 + i]  = (_Float16)b[i];
    f[8 + i]  = (_Float16)c[i];
    f[12 + i] = (_Float16)d[i];
  }
  return f;
}

// A-fragment for the u chunk: ks 0..15 from f32 src (ld=16), ks 16..31 zero.
__device__ __forceinline__ v16h frag_ld16_pad(const float* __restrict__ src,
                                              int ld, int r0, int lane) {
  const float* p = src + (size_t)(r0 + (lane & 15)) * ld + ((lane & 16) ? 8 : 0);
  v4f z = {0.f, 0.f, 0.f, 0.f};
  return cvt16(*(const v4f*)p, *(const v4f*)(p + 4), z, z);
}

// ---------------------------------------------------------------------------
// Fast activations: gfx1250 has V_TANH_F32 (TRANS unit); sigmoid via tanh.
// ---------------------------------------------------------------------------
__device__ __forceinline__ float fast_rcp(float a) {
#if __has_builtin(__builtin_amdgcn_rcpf)
  return __builtin_amdgcn_rcpf(a);
#else
  return 1.f / a;
#endif
}
__device__ __forceinline__ float fast_tanh(float a) {
#if __has_builtin(__builtin_amdgcn_tanhf)
  return __builtin_amdgcn_tanhf(a);
#else
  return 1.f - 2.f * fast_rcp(__expf(2.f * a) + 1.f);
#endif
}
__device__ __forceinline__ float fast_sigmoid(float a) {
#if __has_builtin(__builtin_amdgcn_tanhf)
  return __builtin_amdgcn_tanhf(0.5f * a) * 0.5f + 0.5f;
#else
  return fast_rcp(1.f + __expf(-a));
#endif
}

// ---------------------------------------------------------------------------
// Non-temporal helpers (streaming outputs / read-once inputs).
// v4u is a clang ext-vector (accepted by the nontemporal builtins).
// ---------------------------------------------------------------------------
__device__ __forceinline__ void nt_store_v4u(v4u v, v4u* p) {
#if __has_builtin(__builtin_nontemporal_store)
  __builtin_nontemporal_store(v, p);
#else
  *p = v;
#endif
}
__device__ __forceinline__ void nt_store_f(float v, float* p) {
#if __has_builtin(__builtin_nontemporal_store)
  __builtin_nontemporal_store(v, p);
#else
  *p = v;
#endif
}
__device__ __forceinline__ float nt_load_f(const float* p) {
#if __has_builtin(__builtin_nontemporal_load)
  return __builtin_nontemporal_load(p);
#else
  return *p;
#endif
}

// ---------------------------------------------------------------------------
// Kernel 1a: pack all weights to f16 (MLP W1s K-padded 144 -> 160).
// ---------------------------------------------------------------------------
__global__ __launch_bounds__(256) void dvae_pack(
    const float* __restrict__ dh_wih, const float* __restrict__ dh_whh,
    const float* __restrict__ eh_wih, const float* __restrict__ eh_whh,
    const float* __restrict__ pzw1, const float* __restrict__ pzw2,
    const float* __restrict__ qzw1, const float* __restrict__ qzw2,
    const float* __restrict__ dxw1, const float* __restrict__ dxw2,
    _Float16* __restrict__ w) {
  int i = blockIdx.x * 256 + threadIdx.x;
  if (i >= PACK_TOTAL) return;
  float v;
  if (i < 24576)        v = dh_wih[i];
  else if (i < 73728)   v = dh_whh[i - 24576];
  else if (i < 98304)   v = eh_wih[i - 73728];
  else if (i < 147456)  v = eh_whh[i - 98304];
  else {
    int k = i - 147456;
    if (k < 20480)      { int r = k / 160, c = k % 160;
                          v = (c < 144) ? pzw1[r * 144 + c] : 0.f; }
    else if (k < 28672)   v = pzw2[k - 20480];
    else if (k < 49152) { int k2 = k - 28672; int r = k2 / 160, c = k2 % 160;
                          v = (c < 144) ? qzw1[r * 144 + c] : 0.f; }
    else if (k < 57344)   v = qzw2[k - 49152];
    else if (k < 77824)   v = dxw1[k - 57344];     // 128 x 160 exact
    else                  v = dxw2[k - 77824];     // 128 x 128
  }
  w[i] = (_Float16)v;
}

// ---------------------------------------------------------------------------
// Kernel 1b: pack x to f16 (removes per-step f32->f16 converts from the scan).
// ---------------------------------------------------------------------------
__global__ __launch_bounds__(256) void dvae_pack_x(const float* __restrict__ x,
                                                   _Float16* __restrict__ xh) {
  size_t i = ((size_t)blockIdx.x * 256 + threadIdx.x) * 4;
  v4f a = *(const v4f*)(x + i);
  v4h o;
#pragma unroll
  for (int j = 0; j < 4; ++j) o[j] = (_Float16)a[j];
  *(v4h*)(xh + i) = o;
}

// ---------------------------------------------------------------------------
// Kernel 2: persistent GRU scan.
// ---------------------------------------------------------------------------
__device__ __forceinline__ void h_coop_store(_Float16* __restrict__ hs,
                                             const _Float16* __restrict__ hsm,
                                             int bbase, int t, int tid) {
  int row = tid >> 4;
  int ch  = tid & 15;
  const v4u v = *(const v4u*)&hsm[row * 136 + ch * 8];
  nt_store_v4u(v,
               (v4u*)&hs[((size_t)(bbase + row) * T_DIM + t) * H_DIMC + ch * 8]);
}

__global__ __launch_bounds__(256) void dvae_scan(
    const _Float16* __restrict__ xh, const _Float16* __restrict__ wf,
    const float* __restrict__ dh_bih, const float* __restrict__ dh_bhh,
    const float* __restrict__ eh_bih, const float* __restrict__ eh_bhh,
    _Float16* __restrict__ hs_p, _Float16* __restrict__ hs_q) {
  __shared__ __align__(16) _Float16 hsm[16 * 136];

  const int gru = blockIdx.x >> 3;   // 0: prior d_h (x_{t-1}), 1: e_h (x_t)
  const int bg  = blockIdx.x & 7;
  const _Float16* Wih = wf + (gru ? OFF_EH_WIH : OFF_DH_WIH);
  const _Float16* Whh = wf + (gru ? OFF_EH_WHH : OFF_DH_WHH);
  const float* bih = gru ? eh_bih : dh_bih;
  const float* bhh = gru ? eh_bhh : dh_bhh;
  _Float16* hs = gru ? hs_q : hs_p;

  const int lane = threadIdx.x & 31;
  const int wv   = threadIdx.x >> 5;
  const int j0   = wv * 16;
  const int col  = lane & 15;
  const int rb   = (lane & 16) ? 8 : 0;
  const int j    = j0 + col;
  const int XLD  = T_DIM * X_DIMC;   // row stride of xh in halves

  // Resident weight fragments (loaded once, reused for all 2048 steps).
  v16h wr[4], wz[4], wn[4], uir[2], uiz[2], uin[2];
#pragma unroll
  for (int kc = 0; kc < 4; ++kc) {
    wr[kc] = frag_ld16h(Whh, H_DIMC, j0,              kc * 32, lane);
    wz[kc] = frag_ld16h(Whh, H_DIMC, H_DIMC + j0,     kc * 32, lane);
    wn[kc] = frag_ld16h(Whh, H_DIMC, 2 * H_DIMC + j0, kc * 32, lane);
  }
#pragma unroll
  for (int kc = 0; kc < 2; ++kc) {
    uir[kc] = frag_ld16h(Wih, X_DIMC, j0,              kc * 32, lane);
    uiz[kc] = frag_ld16h(Wih, X_DIMC, H_DIMC + j0,     kc * 32, lane);
    uin[kc] = frag_ld16h(Wih, X_DIMC, 2 * H_DIMC + j0, kc * 32, lane);
  }
  // Biases (r/z merged; n keeps x- and h-side separate: n = tanh(xn + r*hn)).
  const float br  = bih[j] + bhh[j];
  const float bz  = bih[H_DIMC + j] + bhh[H_DIMC + j];
  const float bxn = bih[2 * H_DIMC + j];
  const float bhn = bhh[2 * H_DIMC + j];

  const int bbase = bg * 16;
  float hcur[8];   // this lane's h values (row rb+r, unit j) in f32

  // ---- peeled t = 0 (h_{-1} = 0; prior GRU also has zero x input) ----
  {
    v8f cr = {}, cz = {}, cxn = {};
    if (gru) {   // uniform branch, once
      v16h ax0 = frag_ld16h(xh, XLD, bbase, 0, lane);
      v16h ax1 = frag_ld16h(xh, XLD, bbase, 32, lane);
      cr  = wmma(ax0, uir[0], cr);  cr  = wmma(ax1, uir[1], cr);
      cz  = wmma(ax0, uiz[0], cz);  cz  = wmma(ax1, uiz[1], cz);
      cxn = wmma(ax0, uin[0], cxn); cxn = wmma(ax1, uin[1], cxn);
    }
#pragma unroll
    for (int r = 0; r < 8; ++r) {
      float rg = fast_sigmoid(cr[r] + br);
      float zg = fast_sigmoid(cz[r] + bz);
      float ng = fast_tanh(cxn[r] + bxn + rg * bhn);
      float hn = ng - zg * ng;   // hprev = 0
      hcur[r] = hn;
      hsm[(rb + r) * 136 + j] = (_Float16)hn;
    }
    __syncthreads();
    h_coop_store(hs, hsm, bbase, 0, threadIdx.x);
  }

  // ---- branch-free main loop: t = 1..T-1, tx = t-1+gru always valid ----
  for (int t = 1; t < T_DIM; ++t) {
    const int tx = t - 1 + gru;
    const _Float16* xrow = xh + (size_t)tx * X_DIMC;
    v16h ah[4];
#pragma unroll
    for (int kc = 0; kc < 4; ++kc)
      ah[kc] = frag_ld16h((const _Float16*)hsm, 136, 0, kc * 32, lane);
    v16h ax0 = frag_ld16h(xrow, XLD, bbase, 0, lane);
    v16h ax1 = frag_ld16h(xrow, XLD, bbase, 32, lane);
    __syncthreads();   // all reads of h_{t-1} done before anyone writes h_t

    v8f cr = {}, cz = {}, cxn = {}, chn = {};
#pragma unroll
    for (int kc = 0; kc < 4; ++kc) {
      cr  = wmma(ah[kc], wr[kc], cr);
      cz  = wmma(ah[kc], wz[kc], cz);
      chn = wmma(ah[kc], wn[kc], chn);
    }
    cr  = wmma(ax0, uir[0], cr);  cr  = wmma(ax1, uir[1], cr);
    cz  = wmma(ax0, uiz[0], cz);  cz  = wmma(ax1, uiz[1], cz);
    cxn = wmma(ax0, uin[0], cxn); cxn = wmma(ax1, uin[1], cxn);

#pragma unroll
    for (int r = 0; r < 8; ++r) {
      float rg = fast_sigmoid(cr[r] + br);
      float zg = fast_sigmoid(cz[r] + bz);
      float ng = fast_tanh(cxn[r] + bxn + rg * (chn[r] + bhn));
      float hn = ng + zg * (hcur[r] - ng);   // (1-z)*n + z*h, one FMA
      hcur[r] = hn;
      hsm[(rb + r) * 136 + j] = (_Float16)hn;
    }
    __syncthreads();   // h_t fully in LDS

    h_coop_store(hs, hsm, bbase, t, threadIdx.x);
    if (tx + 1 < T_DIM)
      __builtin_prefetch(
          &xh[((size_t)(bbase + (lane & 15)) * T_DIM + tx + 1) * X_DIMC], 0, 1);
  }
}

// ---------------------------------------------------------------------------
// Kernel 3: pointwise MLPs + z + mu_x + per-block KL partials.
// Rows indexed m = b*T + t so every global access is row-contiguous.
// ---------------------------------------------------------------------------
__global__ __launch_bounds__(256) void dvae_mlp(
    const _Float16* __restrict__ wf,
    const _Float16* __restrict__ hs_p, const _Float16* __restrict__ hs_q,
    const float* __restrict__ u, const float* __restrict__ eps,
    const float* __restrict__ pz_b1, const float* __restrict__ pz_b2,
    const float* __restrict__ qz_b1, const float* __restrict__ qz_b2,
    const float* __restrict__ dx_b1, const float* __restrict__ dx_b2,
    float* __restrict__ out_z, float* __restrict__ out_mux,
    float* __restrict__ kl_partial) {
  extern __shared__ _Float16 smem[];

  {  // stage all six f16 weight matrices (94208 halves) cooperatively
    const _Float16* gw = wf + OFF_MLP;
    for (int i = threadIdx.x; i < MLP_W_HALVES / 8; i += 256)
      *(v4u*)&smem[i * 8] = *(const v4u*)&gw[i * 8];
  }
  __syncthreads();

  const _Float16* sW1P = smem + MO_W1P;
  const _Float16* sW2P = smem + MO_W2P;
  const _Float16* sW1Q = smem + MO_W1Q;
  const _Float16* sW2Q = smem + MO_W2Q;
  const _Float16* sW1D = smem + MO_W1D;
  const _Float16* sW2D = smem + MO_W2D;

  const int lane = threadIdx.x & 31;
  const int wv   = threadIdx.x >> 5;
  const int row0 = blockIdx.x * 128 + wv * 16;   // m = b*T + t
  const int col  = lane & 15;
  const int rb   = (lane & 16) ? 8 : 0;
  _Float16* st = smem + SM_STAGE + wv * (16 * 136);

  v16h ahp[4], ahq[4];
#pragma unroll
  for (int kc = 0; kc < 4; ++kc) {
    ahp[kc] = frag_ld16h(hs_p, H_DIMC, row0, kc * 32, lane);
    ahq[kc] = frag_ld16h(hs_q, H_DIMC, row0, kc * 32, lane);
  }
  v16h au = frag_ld16_pad(u, U_DIMC, row0, lane);

  // ---- p_z MLP ----
#pragma unroll
  for (int nt = 0; nt < 8; ++nt) {
    v8f acc = {};
#pragma unroll
    for (int kc = 0; kc < 4; ++kc)
      acc = wmma(ahp[kc], frag_ld16h(sW1P, 160, nt * 16, kc * 32, lane), acc);
    acc = wmma(au, frag_ld16h(sW1P, 160, nt * 16, 128, lane), acc);
    float b1 = pz_b1[nt * 16 + col];
#pragma unroll
    for (int r = 0; r < 8; ++r) {
      float v = acc[r] + b1;
      st[(rb + r) * 136 + nt * 16 + col] = (_Float16)(v > 0.f ? v : 0.f);
    }
  }
  v16h aa[4];
#pragma unroll
  for (int kc = 0; kc < 4; ++kc)
    aa[kc] = frag_ld16h((const _Float16*)st, 136, 0, kc * 32, lane);
  v8f pacc[4];
#pragma unroll
  for (int nt = 0; nt < 4; ++nt) {
    v8f acc = {};
#pragma unroll
    for (int kc = 0; kc < 4; ++kc)
      acc = wmma(aa[kc], frag_ld16h(sW2P, 128, nt * 16, kc * 32, lane), acc);
    float b2 = pz_b2[nt * 16 + col];
#pragma unroll
    for (int r = 0; r < 8; ++r) acc[r] += b2;
    pacc[nt] = acc;
  }

  // ---- q_z MLP ----
#pragma unroll
  for (int nt = 0; nt < 8; ++nt) {
    v8f acc = {};
#pragma unroll
    for (int kc = 0; kc < 4; ++kc)
      acc = wmma(ahq[kc], frag_ld16h(sW1Q, 160, nt * 16, kc * 32, lane), acc);
    acc = wmma(au, frag_ld16h(sW1Q, 160, nt * 16, 128, lane), acc);
    float b1 = qz_b1[nt * 16 + col];
#pragma unroll
    for (int r = 0; r < 8; ++r) {
      float v = acc[r] + b1;
      st[(rb + r) * 136 + nt * 16 + col] = (_Float16)(v > 0.f ? v : 0.f);
    }
  }
#pragma unroll
  for (int kc = 0; kc < 4; ++kc)
    aa[kc] = frag_ld16h((const _Float16*)st, 136, 0, kc * 32, lane);
  v8f qacc[4];
#pragma unroll
  for (int nt = 0; nt < 4; ++nt) {
    v8f acc = {};
#pragma unroll
    for (int kc = 0; kc < 4; ++kc)
      acc = wmma(aa[kc], frag_ld16h(sW2Q, 128, nt * 16, kc * 32, lane), acc);
    float b2 = qz_b2[nt * 16 + col];
#pragma unroll
    for (int r = 0; r < 8; ++r) acc[r] += b2;
    qacc[nt] = acc;
  }

  // ---- z = mu_q + exp(ls_q)*eps ; KL(q||p) ----
  float klacc = 0.f;
#pragma unroll
  for (int ct = 0; ct < 2; ++ct) {
#pragma unroll
    for (int r = 0; r < 8; ++r) {
      int gr = row0 + rb + r;
      int zc = ct * 16 + col;
      float muq = qacc[ct][r], lsq = qacc[ct + 2][r];
      float mup = pacc[ct][r], lsp = pacc[ct + 2][r];
      float ep = nt_load_f(&eps[(size_t)gr * Z_DIMC + zc]);
      float zv = muq + __expf(lsq) * ep;
      nt_store_f(zv, &out_z[(size_t)gr * Z_DIMC + zc]);
      st[(rb + r) * 136 + zc] = (_Float16)zv;   // z staged for d_x input
      float d = muq - mup;
      klacc += 0.5f * (__expf(2.f * lsq) + d * d) * __expf(-2.f * lsp) +
               (lsp - lsq) - 0.5f;
    }
  }

  // ---- d_x MLP (only mu_x needed -> half of layer 2) ----
  v16h az = frag_ld16h((const _Float16*)st, 136, 0, 0, lane);  // z chunk
#pragma unroll
  for (int nt = 0; nt < 8; ++nt) {
    v8f acc = {};
#pragma unroll
    for (int kc = 0; kc < 4; ++kc)
      acc = wmma(ahp[kc], frag_ld16h(sW1D, 160, nt * 16, kc * 32, lane), acc);
    acc = wmma(az, frag_ld16h(sW1D, 160, nt * 16, 128, lane), acc);
    float b1 = dx_b1[nt * 16 + col];
#pragma unroll
    for (int r = 0; r < 8; ++r) {
      float v = acc[r] + b1;
      st[(rb + r) * 136 + nt * 16 + col] = (_Float16)(v > 0.f ? v : 0.f);
    }
  }
#pragma unroll
  for (int kc = 0; kc < 4; ++kc)
    aa[kc] = frag_ld16h((const _Float16*)st, 136, 0, kc * 32, lane);
#pragma unroll
  for (int nt = 0; nt < 4; ++nt) {   // mu_x = first 64 outputs only
    v8f acc = {};
#pragma unroll
    for (int kc = 0; kc < 4; ++kc)
      acc = wmma(aa[kc], frag_ld16h(sW2D, 128, nt * 16, kc * 32, lane), acc);
    float b2 = dx_b2[nt * 16 + col];
#pragma unroll
    for (int r = 0; r < 8; ++r)
      nt_store_f(acc[r] + b2,
                 &out_mux[(size_t)(row0 + rb + r) * X_DIMC + nt * 16 + col]);
  }

  // ---- deterministic per-block KL partial (fixed summation order) ----
  float* klbuf = (float*)&smem[SM_KL_HALF];
  klbuf[threadIdx.x] = klacc;
  __syncthreads();
  if (threadIdx.x == 0) {
    float s = 0.f;
    for (int i = 0; i < 256; ++i) s += klbuf[i];
    kl_partial[blockIdx.x] = s;
  }
}

// ---------------------------------------------------------------------------
// Kernel 4: final KL reduction (single block, fixed order -> deterministic).
// ---------------------------------------------------------------------------
__global__ __launch_bounds__(256) void dvae_klreduce(
    const float* __restrict__ p, float* __restrict__ out) {
  __shared__ float s[256];
  float a = 0.f;
  for (int i = threadIdx.x; i < MLP_BLOCKS; i += 256) a += p[i];
  s[threadIdx.x] = a;
  __syncthreads();
  if (threadIdx.x == 0) {
    float t = 0.f;
    for (int i = 0; i < 256; ++i) t += s[i];
    *out = t / (float)(B_DIM * T_DIM);
  }
}

// ---------------------------------------------------------------------------
extern "C" void kernel_launch(void* const* d_in, const int* in_sizes, int n_in,
                              void* d_out, int out_size, void* d_ws,
                              size_t ws_size, hipStream_t stream) {
  const float* x      = (const float*)d_in[0];
  const float* u      = (const float*)d_in[1];
  const float* eps    = (const float*)d_in[2];
  const float* dh_wih = (const float*)d_in[3];
  const float* dh_whh = (const float*)d_in[4];
  const float* dh_bih = (const float*)d_in[5];
  const float* dh_bhh = (const float*)d_in[6];
  const float* eh_wih = (const float*)d_in[7];
  const float* eh_whh = (const float*)d_in[8];
  const float* eh_bih = (const float*)d_in[9];
  const float* eh_bhh = (const float*)d_in[10];
  const float* pz_w1  = (const float*)d_in[11];
  const float* pz_b1  = (const float*)d_in[12];
  const float* pz_w2  = (const float*)d_in[13];
  const float* pz_b2  = (const float*)d_in[14];
  const float* qz_w1  = (const float*)d_in[15];
  const float* qz_b1  = (const float*)d_in[16];
  const float* qz_w2  = (const float*)d_in[17];
  const float* qz_b2  = (const float*)d_in[18];
  const float* dx_w1  = (const float*)d_in[19];
  const float* dx_b1  = (const float*)d_in[20];
  const float* dx_w2  = (const float*)d_in[21];
  const float* dx_b2  = (const float*)d_in[22];

  _Float16* wf   = (_Float16*)d_ws;
  _Float16* hs_p = (_Float16*)((char*)d_ws + WS_HSP_OFF);
  _Float16* hs_q = (_Float16*)((char*)d_ws + WS_HSQ_OFF);
  float*    klp  = (float*)((char*)d_ws + WS_KLP_OFF);
  _Float16* xh   = (_Float16*)((char*)d_ws + WS_XH_OFF);

  float* out_z   = (float*)d_out;
  float* out_mux = out_z + (size_t)B_DIM * T_DIM * Z_DIMC;
  float* out_kl  = out_z + (size_t)B_DIM * T_DIM * (Z_DIMC + X_DIMC);

  dvae_pack<<<dim3((PACK_TOTAL + 255) / 256), dim3(256), 0, stream>>>(
      dh_wih, dh_whh, eh_wih, eh_whh, pz_w1, pz_w2, qz_w1, qz_w2, dx_w1, dx_w2,
      wf);
  dvae_pack_x<<<dim3((unsigned)(X_ELEMS / 4 / 256)), dim3(256), 0, stream>>>(
      x, xh);

  dvae_scan<<<dim3(16), dim3(256), 0, stream>>>(
      xh, wf, dh_bih, dh_bhh, eh_bih, eh_bhh, hs_p, hs_q);

  (void)hipFuncSetAttribute(reinterpret_cast<const void*>(dvae_mlp),
                            hipFuncAttributeMaxDynamicSharedMemorySize,
                            MLP_SMEM_BYTES);
  dvae_mlp<<<dim3(MLP_BLOCKS), dim3(256), MLP_SMEM_BYTES, stream>>>(
      wf, hs_p, hs_q, u, eps, pz_b1, pz_b2, qz_b1, qz_b2, dx_b1, dx_b2, out_z,
      out_mux, klp);

  dvae_klreduce<<<dim3(1), dim3(256), 0, stream>>>(klp, out_kl);
}
